// HeteroGNN_23330262352208
// MI455X (gfx1250) — matrix operand order repaired and checked by reading, not compiled, not run on hardware
//
#include <hip/hip_runtime.h>
#include <hip/hip_bf16.h>
#include <math.h>

#define NG   100000
#define ND   20000
#define FIN  128
#define HID  128
#define FOUT 64
#define EGG  400000
#define EGD  100000
#define EDG  100000
#define SLOPE 0.2f

typedef __attribute__((ext_vector_type(16))) _Float16 v16h;
typedef __attribute__((ext_vector_type(8)))  float    v8f;

// ---------------------------------------------------------------- utilities

__global__ void k_f32_to_f16(const float* __restrict__ x, _Float16* __restrict__ y, int n) {
    int i = blockIdx.x * blockDim.x + threadIdx.x;
    if (i < n) y[i] = (_Float16)x[i];
}

// BT[n*K + k] = (f16) W[k*N + n]   (weights -> transposed f16 for GEMM B operand)
__global__ void k_f32_to_f16_T(const float* __restrict__ W, _Float16* __restrict__ BT,
                               int K, int N) {
    int i = blockIdx.x * blockDim.x + threadIdx.x;
    if (i >= K * N) return;
    int n = i / K, k = i - n * K;
    BT[i] = (_Float16)W[k * N + n];
}

__global__ void k_fill(float* __restrict__ x, float v, int n) {
    int i = blockIdx.x * blockDim.x + threadIdx.x;
    if (i < n) x[i] = v;
}

// v[t][k] = sum_n W[t][k][n] * a[t][n]   (collapses (x@Wd)@a_d -> x . v)
__global__ void k_matvec_wv(const float* __restrict__ W, const float* __restrict__ a,
                            float* __restrict__ v) {
    int t = blockIdx.x;
    int k = threadIdx.x;
    const float* Wr = W + t * HID * HID + k * HID;
    const float* ar = a + t * HID;
    float s = 0.f;
#pragma unroll 8
    for (int n = 0; n < HID; ++n) s += Wr[n] * ar[n];
    v[t * HID + k] = s;
}

// e[i] = dot(X[i,:], v)  (K = 128)
__global__ void k_rowdot(const float* __restrict__ X, const float* __restrict__ v,
                         float* __restrict__ e, int M) {
    int i = blockIdx.x * blockDim.x + threadIdx.x;
    if (i >= M) return;
    const float* x = X + (size_t)i * HID;
    float s = 0.f;
#pragma unroll 8
    for (int k = 0; k < HID; ++k) s += x[k] * v[k];
    e[i] = s;
}

// out[i*H + h] = b0[h] (+ b1[h])
__global__ void k_bias_init(float* __restrict__ out, const float* __restrict__ b0,
                            const float* __restrict__ b1, int M) {
    int i = blockIdx.x * blockDim.x + threadIdx.x;
    if (i >= M * HID) return;
    int h = i & (HID - 1);
    float v = b0[h];
    if (b1) v += b1[h];
    out[i] = v;
}

__device__ __forceinline__ void atomicMaxF(float* addr, float val) {
    unsigned int* u = (unsigned int*)addr;
    unsigned int old = *u;
    while (__uint_as_float(old) < val) {
        unsigned int assumed = old;
        old = atomicCAS(u, assumed, __float_as_uint(val));
        if (old == assumed) break;
    }
}

// ---------------------------------------------------------------- edge softmax

__global__ void k_edge_max(const int* __restrict__ src, const int* __restrict__ dst,
                           const float* __restrict__ es, const float* __restrict__ ed,
                           float* __restrict__ e, float* __restrict__ m, int E) {
    int i = blockIdx.x * blockDim.x + threadIdx.x;
    if (i >= E) return;
    float v = es[src[i]] + ed[dst[i]];
    v = (v > 0.f) ? v : SLOPE * v;
    e[i] = v;
    atomicMaxF(&m[dst[i]], v);
}

__global__ void k_edge_exp(const int* __restrict__ dst, const float* __restrict__ m,
                           float* __restrict__ e, float* __restrict__ den, int E) {
    int i = blockIdx.x * blockDim.x + threadIdx.x;
    if (i >= E) return;
    int d = dst[i];
    float ex = __expf(e[i] - m[d]);
    e[i] = ex;
    atomicAdd(&den[d], ex);
}

// one wave (32 lanes) per edge, 4 features per lane (H=128)
__global__ void k_edge_scatter(const int* __restrict__ src, const int* __restrict__ dst,
                               const float* __restrict__ ex, const float* __restrict__ den,
                               const float* __restrict__ hs, float* __restrict__ out, int E) {
    int wid  = threadIdx.x >> 5;
    int lane = threadIdx.x & 31;
    int i = blockIdx.x * 8 + wid;
    if (i >= E) return;
    int s = src[i], d = dst[i];
    float alpha = ex[i] / den[d];
    const float* h = hs + (size_t)s * HID;
    float* o = out + (size_t)d * HID;
#pragma unroll
    for (int q = 0; q < 4; ++q) {
        int f = lane + q * 32;
        atomicAdd(&o[f], alpha * h[f]);
    }
}

__global__ void k_relu_f16(float* __restrict__ x, _Float16* __restrict__ y, int n) {
    int i = blockIdx.x * blockDim.x + threadIdx.x;
    if (i >= n) return;
    float v = x[i];
    v = v > 0.f ? v : 0.f;
    x[i] = v;
    y[i] = (_Float16)v;
}

__global__ void k_add_bias(float* __restrict__ out, const float* __restrict__ b, int M, int N) {
    int i = blockIdx.x * blockDim.x + threadIdx.x;
    if (i >= M * N) return;
    out[i] += b[i % N];
}

// ---------------------------------------------------------------- WMMA GEMM
// C[M, NT*16] = A[M,128] @ B[128, NT*16].
// A: f16 row-major. BT: f16 TRANSPOSED weights [N][K=128] (contiguous K per column),
// staged in LDS once per block. Each wave computes a 32-row strip (two 16-row
// accumulator tiles per column tile): each LDS B fragment feeds 2 WMMAs, and B
// fragments are double-buffered so ds_load latency overlaps the XDL pipe.
// __launch_bounds__(256, 1): allow ~190 VGPRs/wave so all 16 accumulator tiles
// stay register-resident (default 128-VGPR budget caused scratch spills).

template <int NT>
__global__ __launch_bounds__(256, 1)
void k_gemm_wmma(const _Float16* __restrict__ A, const _Float16* __restrict__ BT,
                 float* __restrict__ C, int M) {
    constexpr int K = 128;
    constexpr int N = NT * 16;
    __shared__ _Float16 sB[N * K];

    // cooperative 16B-chunk copy of BT into LDS
    for (int c = threadIdx.x; c < (N * K) / 8; c += blockDim.x)
        ((uint4*)sB)[c] = ((const uint4*)BT)[c];
    __syncthreads();

    const int wid  = threadIdx.x >> 5;
    const int lane = threadIdx.x & 31;
    const int pair = blockIdx.x * 8 + wid;           // 32-row block (wave-uniform)
    if (pair >= (M >> 5)) return;                    // after barrier: safe

    const int half = lane >> 4;
    const int r    = lane & 15;
    const _Float16* Arow0 = A + (size_t)((pair << 5) + r) * K;
    const _Float16* Arow1 = Arow0 + (size_t)16 * K;
    const _Float16* Bbase = sB + (size_t)r * K + half * 16;   // column nt*16+r

    union F16x16 { v16h v; uint4 u[2]; };

    v8f acc0[NT], acc1[NT];
#pragma unroll
    for (int nt = 0; nt < NT; ++nt) {
        acc0[nt] = (v8f){0.f, 0.f, 0.f, 0.f, 0.f, 0.f, 0.f, 0.f};
        acc1[nt] = (v8f){0.f, 0.f, 0.f, 0.f, 0.f, 0.f, 0.f, 0.f};
    }

#pragma unroll
    for (int k0 = 0; k0 < K; k0 += 32) {
        // A 16x32 f16 fragments: lane half picks K = 16*half+0..15
        F16x16 a0, a1;
        a0.u[0] = *(const uint4*)(Arow0 + k0 + half * 16);
        a0.u[1] = *(const uint4*)(Arow0 + k0 + half * 16 + 8);
        a1.u[0] = *(const uint4*)(Arow1 + k0 + half * 16);
        a1.u[1] = *(const uint4*)(Arow1 + k0 + half * 16 + 8);

        // B 32x16 fragment: column nt*16+r, K = 16*half+0..15 (contiguous in BT)
        F16x16 bcur, bnext;
        bcur.u[0] = *(const uint4*)(Bbase + k0);
        bcur.u[1] = *(const uint4*)(Bbase + k0 + 8);
#pragma unroll
        for (int nt = 0; nt < NT; ++nt) {
            if (nt + 1 < NT) {  // prefetch next B fragment before using current
                const _Float16* bp = Bbase + (size_t)(nt + 1) * 16 * K + k0;
                bnext.u[0] = *(const uint4*)(bp);
                bnext.u[1] = *(const uint4*)(bp + 8);
            }
            acc0[nt] = __builtin_amdgcn_wmma_f32_16x16x32_f16(
                false, a0.v, false, bcur.v, (short)0, acc0[nt], false, false);
            acc1[nt] = __builtin_amdgcn_wmma_f32_16x16x32_f16(
                false, a1.v, false, bcur.v, (short)0, acc1[nt], false, false);
            if (nt + 1 < NT) bcur = bnext;
        }
    }

    // C layout: VGPR rr -> M = mbase + rr, N = nt*16 + (lane&15)
    const int mbase0 = (pair << 5) + half * 8;
    const int mbase1 = mbase0 + 16;
#pragma unroll
    for (int nt = 0; nt < NT; ++nt) {
        float* c0 = C + (size_t)mbase0 * N + nt * 16 + r;
        float* c1 = C + (size_t)mbase1 * N + nt * 16 + r;
#pragma unroll
        for (int rr = 0; rr < 8; ++rr) {
            c0[(size_t)rr * N] = acc0[nt][rr];
            c1[(size_t)rr * N] = acc1[nt][rr];
        }
    }
}

// ---------------------------------------------------------------- host side

static inline int cdiv(int a, int b) { return (a + b - 1) / b; }

static void gat_edges(const int* src, const int* dst, int E,
                      const float* es, const float* ed, const float* hs,
                      float* out, int n_dst,
                      float* mbuf, float* denbuf, float* ebuf, hipStream_t st) {
    k_fill<<<cdiv(n_dst, 256), 256, 0, st>>>(mbuf, -INFINITY, n_dst);
    k_fill<<<cdiv(n_dst, 256), 256, 0, st>>>(denbuf, 0.f, n_dst);
    k_edge_max<<<cdiv(E, 256), 256, 0, st>>>(src, dst, es, ed, ebuf, mbuf, E);
    k_edge_exp<<<cdiv(E, 256), 256, 0, st>>>(dst, mbuf, ebuf, denbuf, E);
    k_edge_scatter<<<cdiv(E, 8), 256, 0, st>>>(src, dst, ebuf, denbuf, hs, out, E);
}

static void gemm128(const _Float16* A, const _Float16* BT, float* C, int M, int N,
                    hipStream_t st) {
    int blocks = cdiv(M >> 5, 8);   // 32-row strips, 8 waves per block
    if (N == 128) k_gemm_wmma<8><<<blocks, 256, 0, st>>>(A, BT, C, M);
    else          k_gemm_wmma<4><<<blocks, 256, 0, st>>>(A, BT, C, M);
}

extern "C" void kernel_launch(void* const* d_in, const int* in_sizes, int n_in,
                              void* d_out, int out_size, void* d_ws, size_t ws_size,
                              hipStream_t stream) {
    (void)in_sizes; (void)n_in; (void)out_size; (void)ws_size;

    const float* x_gene    = (const float*)d_in[0];
    const float* x_disease = (const float*)d_in[1];
    const float* Wsrc1 = (const float*)d_in[2];
    const float* Wdst1 = (const float*)d_in[3];
    const float* atts1 = (const float*)d_in[4];
    const float* attd1 = (const float*)d_in[5];
    const float* b1    = (const float*)d_in[6];
    const float* Wsrc2 = (const float*)d_in[7];
    const float* Wdst2 = (const float*)d_in[8];
    const float* atts2 = (const float*)d_in[9];
    const float* attd2 = (const float*)d_in[10];
    const float* b2    = (const float*)d_in[11];
    const float* Wg = (const float*)d_in[12];
    const float* bg = (const float*)d_in[13];
    const float* Wd = (const float*)d_in[14];
    const float* bd = (const float*)d_in[15];
    const int* g2g_src = (const int*)d_in[16];
    const int* g2g_dst = (const int*)d_in[17];
    const int* g2d_src = (const int*)d_in[18];
    const int* g2d_dst = (const int*)d_in[19];
    const int* d2g_src = (const int*)d_in[20];
    const int* d2g_dst = (const int*)d_in[21];
    float* out = (float*)d_out;

    // -------- workspace bump allocator (256B aligned)
    char* p = (char*)d_ws;
    auto alloc = [&](size_t bytes) -> void* {
        void* r = (void*)p;
        p += (bytes + 255) & ~(size_t)255;
        return r;
    };
    _Float16* xg16  = (_Float16*)alloc((size_t)NG * HID * 2);
    _Float16* xd16  = (_Float16*)alloc((size_t)ND * HID * 2);
    _Float16* w1_16 = (_Float16*)alloc((size_t)3 * HID * HID * 2);  // transposed slices
    _Float16* w2_16 = (_Float16*)alloc((size_t)3 * HID * HID * 2);  // transposed slices
    _Float16* wg16  = (_Float16*)alloc((size_t)HID * FOUT * 2);     // transposed
    _Float16* wd16  = (_Float16*)alloc((size_t)HID * FOUT * 2);     // transposed
    float* hs_gg = (float*)alloc((size_t)NG * HID * 4);
    float* hs_gd = (float*)alloc((size_t)NG * HID * 4);
    float* hs_dg = (float*)alloc((size_t)ND * HID * 4);
    float* og    = (float*)alloc((size_t)NG * HID * 4);
    float* od    = (float*)alloc((size_t)ND * HID * 4);
    float* og2   = (float*)alloc((size_t)NG * HID * 4);
    float* od2   = (float*)alloc((size_t)ND * HID * 4);
    float* es_gg = (float*)alloc((size_t)NG * 4);
    float* es_gd = (float*)alloc((size_t)NG * 4);
    float* es_dg = (float*)alloc((size_t)ND * 4);
    float* ed_gg = (float*)alloc((size_t)NG * 4);
    float* ed_gd = (float*)alloc((size_t)ND * 4);
    float* ed_dg = (float*)alloc((size_t)NG * 4);
    float* vdst  = (float*)alloc((size_t)3 * HID * 4);
    float* mbuf  = (float*)alloc((size_t)NG * 4);
    float* denb  = (float*)alloc((size_t)NG * 4);
    float* ebuf  = (float*)alloc((size_t)EGG * 4);

    // -------- operand conversions (A row-major f16, weights transposed f16)
    k_f32_to_f16<<<cdiv(NG * HID, 256), 256, 0, stream>>>(x_gene, xg16, NG * HID);
    k_f32_to_f16<<<cdiv(ND * HID, 256), 256, 0, stream>>>(x_disease, xd16, ND * HID);
    for (int t = 0; t < 3; ++t) {
        k_f32_to_f16_T<<<cdiv(HID * HID, 256), 256, 0, stream>>>(
            Wsrc1 + t * HID * HID, w1_16 + t * HID * HID, HID, HID);
        k_f32_to_f16_T<<<cdiv(HID * HID, 256), 256, 0, stream>>>(
            Wsrc2 + t * HID * HID, w2_16 + t * HID * HID, HID, HID);
    }
    k_f32_to_f16_T<<<cdiv(HID * FOUT, 256), 256, 0, stream>>>(Wg, wg16, HID, FOUT);
    k_f32_to_f16_T<<<cdiv(HID * FOUT, 256), 256, 0, stream>>>(Wd, wd16, HID, FOUT);

    // ================= LAYER 1 =================
    k_matvec_wv<<<3, HID, 0, stream>>>(Wdst1, attd1, vdst);
    gemm128(xg16, w1_16,                 hs_gg, NG, HID, stream);
    gemm128(xg16, w1_16 + HID * HID,     hs_gd, NG, HID, stream);
    gemm128(xd16, w1_16 + 2 * HID * HID, hs_dg, ND, HID, stream);
    k_rowdot<<<cdiv(NG, 256), 256, 0, stream>>>(hs_gg, atts1,           es_gg, NG);
    k_rowdot<<<cdiv(NG, 256), 256, 0, stream>>>(hs_gd, atts1 + HID,     es_gd, NG);
    k_rowdot<<<cdiv(ND, 256), 256, 0, stream>>>(hs_dg, atts1 + 2 * HID, es_dg, ND);
    k_rowdot<<<cdiv(NG, 256), 256, 0, stream>>>(x_gene,    vdst,           ed_gg, NG);
    k_rowdot<<<cdiv(ND, 256), 256, 0, stream>>>(x_disease, vdst + HID,     ed_gd, ND);
    k_rowdot<<<cdiv(NG, 256), 256, 0, stream>>>(x_gene,    vdst + 2 * HID, ed_dg, NG);
    k_bias_init<<<cdiv(NG * HID, 256), 256, 0, stream>>>(og, b1, b1 + 2 * HID, NG);
    k_bias_init<<<cdiv(ND * HID, 256), 256, 0, stream>>>(od, b1 + HID, nullptr, ND);
    gat_edges(g2g_src, g2g_dst, EGG, es_gg, ed_gg, hs_gg, og, NG, mbuf, denb, ebuf, stream);
    gat_edges(d2g_src, d2g_dst, EDG, es_dg, ed_dg, hs_dg, og, NG, mbuf, denb, ebuf, stream);
    gat_edges(g2d_src, g2d_dst, EGD, es_gd, ed_gd, hs_gd, od, ND, mbuf, denb, ebuf, stream);
    k_relu_f16<<<cdiv(NG * HID, 256), 256, 0, stream>>>(og, xg16, NG * HID);
    k_relu_f16<<<cdiv(ND * HID, 256), 256, 0, stream>>>(od, xd16, ND * HID);

    // ================= LAYER 2 =================
    k_matvec_wv<<<3, HID, 0, stream>>>(Wdst2, attd2, vdst);
    gemm128(xg16, w2_16,                 hs_gg, NG, HID, stream);
    gemm128(xg16, w2_16 + HID * HID,     hs_gd, NG, HID, stream);
    gemm128(xd16, w2_16 + 2 * HID * HID, hs_dg, ND, HID, stream);
    k_rowdot<<<cdiv(NG, 256), 256, 0, stream>>>(hs_gg, atts2,           es_gg, NG);
    k_rowdot<<<cdiv(NG, 256), 256, 0, stream>>>(hs_gd, atts2 + HID,     es_gd, NG);
    k_rowdot<<<cdiv(ND, 256), 256, 0, stream>>>(hs_dg, atts2 + 2 * HID, es_dg, ND);
    k_rowdot<<<cdiv(NG, 256), 256, 0, stream>>>(og, vdst,           ed_gg, NG);
    k_rowdot<<<cdiv(ND, 256), 256, 0, stream>>>(od, vdst + HID,     ed_gd, ND);
    k_rowdot<<<cdiv(NG, 256), 256, 0, stream>>>(og, vdst + 2 * HID, ed_dg, NG);
    k_bias_init<<<cdiv(NG * HID, 256), 256, 0, stream>>>(og2, b2, b2 + 2 * HID, NG);
    k_bias_init<<<cdiv(ND * HID, 256), 256, 0, stream>>>(od2, b2 + HID, nullptr, ND);
    gat_edges(g2g_src, g2g_dst, EGG, es_gg, ed_gg, hs_gg, og2, NG, mbuf, denb, ebuf, stream);
    gat_edges(d2g_src, d2g_dst, EDG, es_dg, ed_dg, hs_dg, og2, NG, mbuf, denb, ebuf, stream);
    gat_edges(g2d_src, g2d_dst, EGD, es_gd, ed_gd, hs_gd, od2, ND, mbuf, denb, ebuf, stream);
    k_relu_f16<<<cdiv(NG * HID, 256), 256, 0, stream>>>(og2, xg16, NG * HID);
    k_relu_f16<<<cdiv(ND * HID, 256), 256, 0, stream>>>(od2, xd16, ND * HID);

    // ================= output heads =================
    gemm128(xg16, wg16, out, NG, FOUT, stream);
    k_add_bias<<<cdiv(NG * FOUT, 256), 256, 0, stream>>>(out, bg, NG, FOUT);
    gemm128(xd16, wd16, out + (size_t)NG * FOUT, ND, FOUT, stream);
    k_add_bias<<<cdiv(ND * FOUT, 256), 256, 0, stream>>>(out + (size_t)NG * FOUT, bd, ND, FOUT);
}
